// BLCD_Loss_61959198212393
// MI455X (gfx1250) — compile-verified, matrix operand order checked by reading, not run on hardware
//
#include <hip/hip_runtime.h>
#include <hip/hip_bf16.h>
#include <math.h>

#define NPTS 8192
#define DIM  128
#define KNN  16
#define EPS_F 1e-12f
#define T_F   0.0025f
#define M_F   0.5f

typedef __attribute__((ext_vector_type(16))) _Float16 v16h;
typedef __attribute__((ext_vector_type(8)))  _Float16 v8h;
typedef __attribute__((ext_vector_type(4)))  _Float16 v4h;
typedef __attribute__((ext_vector_type(8)))  float    v8f;

// ---------------------------------------------------------------------------
// Fragment loader for V_WMMA_F32_16X16X32_F16 on row-major f16 data.
// A (16x32, M x K) and B (32x16, K x N) use the same per-lane pattern when B is
// taken from row-major storage indexed by column (i.e. dot-product / A*A^T):
//   lanes 0-15  : row r, halves [k0..k0+7]  and [k0+16..k0+23]
//   lanes 16-31 : row r, halves [k0+8..k0+15] and [k0+24..k0+31]
// ---------------------------------------------------------------------------
__device__ inline v16h load_frag(const _Float16* __restrict__ base, int row,
                                 int k0, int laneHi) {
    const _Float16* p = base + row * DIM + k0 + laneHi * 8;
    v8h lo = *(const v8h*)(p);
    v8h hi = *(const v8h*)(p + 16);
    return __builtin_shufflevector(lo, hi, 0, 1, 2, 3, 4, 5, 6, 7,
                                           8, 9, 10, 11, 12, 13, 14, 15);
}

__device__ inline v8f wmma_f16(v16h a, v16h b, v8f c) {
    return __builtin_amdgcn_wmma_f32_16x16x32_f16(
        /*neg_a=*/false, a, /*neg_b=*/false, b,
        /*c_mod=*/(short)0, c, /*reuse_a=*/false, /*reuse_b=*/false);
}

__device__ inline float wave_sum(float s) {
    #pragma unroll
    for (int o = 16; o > 0; o >>= 1) s += __shfl_xor(s, o, 32);
    return s;
}

// ---------------------------------------------------------------------------
// K1: L2-normalize yi -> yinF (f32) + yinH (f16) + sq;  yi_t -> yitF (f32).
// One wave per row; lane handles 4 contiguous dims.
// ---------------------------------------------------------------------------
__global__ __launch_bounds__(256) void normalize_kernel(
    const float* __restrict__ yi, const float* __restrict__ yt,
    float* __restrict__ yinF, float* __restrict__ yitF,
    _Float16* __restrict__ yinH, float* __restrict__ sqArr) {
    int lane = threadIdx.x & 31;
    int row  = blockIdx.x * 8 + (threadIdx.x >> 5);

    const float4 x = ((const float4*)(yi + (size_t)row * DIM))[lane];
    float s = wave_sum(x.x * x.x + x.y * x.y + x.z * x.z + x.w * x.w);
    float inv = 1.0f / sqrtf(s + EPS_F);
    float4 nx = make_float4(x.x * inv, x.y * inv, x.z * inv, x.w * inv);
    ((float4*)(yinF + (size_t)row * DIM))[lane] = nx;
    v4h h = { (_Float16)nx.x, (_Float16)nx.y, (_Float16)nx.z, (_Float16)nx.w };
    *(v4h*)(yinH + (size_t)row * DIM + lane * 4) = h;
    // recompute sq = sum(yin^2) after normalization (matches reference)
    float s2 = wave_sum(nx.x * nx.x + nx.y * nx.y + nx.z * nx.z + nx.w * nx.w);
    if (lane == 0) sqArr[row] = s2;

    const float4 t = ((const float4*)(yt + (size_t)row * DIM))[lane];
    float st = wave_sum(t.x * t.x + t.y * t.y + t.z * t.z + t.w * t.w);
    float invt = 1.0f / sqrtf(st + EPS_F);
    float4 nt = make_float4(t.x * invt, t.y * invt, t.z * invt, t.w * invt);
    ((float4*)(yitF + (size_t)row * DIM))[lane] = nt;
}

// ---------------------------------------------------------------------------
// K2: fused WMMA similarity GEMM + stable per-row top-17 (smallest key).
// key(i,j) = sq_j - 2*dot(i,j)   (order-equivalent to the reference distance).
// Workgroup = 16 waves, handles 16 rows.  Per 512-col strip: each wave does
// two 16x16 tiles (4 x v_wmma_f32_16x16x32_f16), dumps keys to LDS, then each
// wave merges one row's 512 candidates into a register-sorted top-17 list
// (entry e lives in lane e; lane 16 = current threshold).
// ---------------------------------------------------------------------------
#define STRIP_COLS 512
#define STRIP_LD   516   // padded f32 stride (kills lane0-15/16-31 bank alias)

__global__ __launch_bounds__(512) void topk_kernel(
    const _Float16* __restrict__ yinH, const float* __restrict__ sqArr,
    int* __restrict__ topIdx) {
    __shared__ float strip[16 * STRIP_LD];

    const int tid = threadIdx.x;
    const int lane = tid & 31, wave = tid >> 5;
    const int laneRow = lane & 15, laneHi = lane >> 4;
    const int i0 = blockIdx.x * 16;

    // A fragments for this 16-row block, K = 128 in 4 k-steps
    const v16h a0 = load_frag(yinH, i0 + laneRow,  0, laneHi);
    const v16h a1 = load_frag(yinH, i0 + laneRow, 32, laneHi);
    const v16h a2 = load_frag(yinH, i0 + laneRow, 64, laneHi);
    const v16h a3 = load_frag(yinH, i0 + laneRow, 96, laneHi);

    // register-resident sorted list: lane e holds e-th smallest (e = 0..16)
    float listVal = INFINITY;
    int   listIdx = -1;

    for (int s = 0; s < NPTS / STRIP_COLS; ++s) {
        const int jbase = s * STRIP_COLS;

        // ---- GEMM phase: this wave computes tiles (2*wave) and (2*wave+1)
        #pragma unroll
        for (int t = 0; t < 2; ++t) {
            const int tile = wave * 2 + t;
            const int j0 = jbase + tile * 16;
            v8f c = {};
            { v16h b = load_frag(yinH, j0 + laneRow,  0, laneHi); c = wmma_f16(a0, b, c); }
            { v16h b = load_frag(yinH, j0 + laneRow, 32, laneHi); c = wmma_f16(a1, b, c); }
            { v16h b = load_frag(yinH, j0 + laneRow, 64, laneHi); c = wmma_f16(a2, b, c); }
            { v16h b = load_frag(yinH, j0 + laneRow, 96, laneHi); c = wmma_f16(a3, b, c); }
            const float sqj = sqArr[j0 + laneRow];
            #pragma unroll
            for (int e = 0; e < 8; ++e) {
                const int m = laneHi * 8 + e;  // C-layout: vgpr e, lanes0-15->M=e, 16-31->M=e+8
                strip[m * STRIP_LD + tile * 16 + laneRow] = sqj - 2.0f * c[e];
            }
        }
        __syncthreads();

        // ---- selection phase: wave w owns row w
        float thresh = __shfl(listVal, 16, 32);
        #pragma unroll 1
        for (int ch = 0; ch < STRIP_COLS / 32; ++ch) {
            const int lc = ch * 32 + lane;
            const float v = strip[wave * STRIP_LD + lc];
            const int   j = jbase + lc;
            unsigned mask = (unsigned)__ballot(v < thresh);
            while (mask) {
                const int src = __builtin_ctz(mask);
                mask &= mask - 1;
                const float cv = __shfl(v, src, 32);
                const int   cj = __shfl(j, src, 32);
                if (cv < thresh) {   // re-check against updated threshold
                    // parallel sorted insert (stable: strict > keeps earlier ties first)
                    const bool  myGt = listVal > cv;
                    const float pv = __shfl_up(listVal, 1, 32);
                    const int   pi = __shfl_up(listIdx, 1, 32);
                    const bool  pGt = (lane > 0) && (pv > cv);
                    listVal = myGt ? (pGt ? pv : cv) : listVal;
                    listIdx = myGt ? (pGt ? pi : cj) : listIdx;
                    thresh = __shfl(listVal, 16, 32);
                }
            }
        }
        __syncthreads();
    }

    // lane 0 = self (key ~ -1); lanes 1..16 = the 16 nearest neighbors
    if (lane >= 1 && lane <= KNN)
        topIdx[(size_t)(i0 + wave) * KNN + (lane - 1)] = listIdx;
}

// ---------------------------------------------------------------------------
// K3: per-row loss terms; deterministic block-level partial sums.
// ---------------------------------------------------------------------------
__global__ __launch_bounds__(256) void loss_kernel(
    const float* __restrict__ yinF, const float* __restrict__ yitF,
    const int* __restrict__ topIdx, float* __restrict__ partial) {
    __shared__ float se1[8], se2[8];
    const int lane = threadIdx.x & 31, wave = threadIdx.x >> 5;
    const int row = blockIdx.x * 8 + wave;

    const float4 xi = ((const float4*)(yinF + (size_t)row * DIM))[lane];
    const float4 ti = ((const float4*)(yitF + (size_t)row * DIM))[lane];

    float dx = xi.x - ti.x, dy = xi.y - ti.y, dz = xi.z - ti.z, dw = xi.w - ti.w;
    const float dt = wave_sum(dx * dx + dy * dy + dz * dz + dw * dw);

    float e1 = 0.0f, d0 = 0.0f;
    #pragma unroll 1
    for (int k = 0; k < KNN; ++k) {
        const int j = topIdx[(size_t)row * KNN + k];
        const float4 yj = ((const float4*)(yinF + (size_t)j * DIM))[lane];
        float ax = xi.x - yj.x, ay = xi.y - yj.y, az = xi.z - yj.z, aw = xi.w - yj.w;
        float bx = ti.x - yj.x, by = ti.y - yj.y, bz = ti.z - yj.z, bw = ti.w - yj.w;
        float a = ax * ax + ay * ay + az * az + aw * aw;
        float b = bx * bx + by * by + bz * bz + bw * bw;
        #pragma unroll
        for (int o = 16; o > 0; o >>= 1) { a += __shfl_xor(a, o, 32); b += __shfl_xor(b, o, 32); }
        const float dk  = 0.5f * sqrtf(a + EPS_F);
        const float dtk = 0.5f * sqrtf(b + EPS_F);
        const float o1 = (dk - dtk) * (dk - dtk) - T_F;
        e1 += fmaxf(o1, 0.0f);
        if (k == 0) d0 = dk;
    }
    const float e2 = fmaxf(0.5f * sqrtf(dt + EPS_F) + M_F - d0, 0.0f);

    if (lane == 0) { se1[wave] = e1; se2[wave] = e2; }
    __syncthreads();
    if (threadIdx.x == 0) {
        float a = 0.0f, b = 0.0f;
        #pragma unroll
        for (int w = 0; w < 8; ++w) { a += se1[w]; b += se2[w]; }
        partial[blockIdx.x * 2 + 0] = a;
        partial[blockIdx.x * 2 + 1] = b;
    }
}

// ---------------------------------------------------------------------------
// K4: deterministic final tree reduction of 1024 partial pairs.
// ---------------------------------------------------------------------------
__global__ __launch_bounds__(512) void finalize_kernel(
    const float* __restrict__ partial, float* __restrict__ out) {
    __shared__ float s1[512], s2[512];
    const int t = threadIdx.x;
    s1[t] = partial[t * 2 + 0] + partial[(t + 512) * 2 + 0];
    s2[t] = partial[t * 2 + 1] + partial[(t + 512) * 2 + 1];
    __syncthreads();
    #pragma unroll 1
    for (int o = 256; o > 0; o >>= 1) {
        if (t < o) { s1[t] += s1[t + o]; s2[t] += s2[t + o]; }
        __syncthreads();
    }
    if (t == 0) { out[0] = s1[0] + s2[0]; out[1] = s1[0]; out[2] = s2[0]; }
}

// ---------------------------------------------------------------------------
extern "C" void kernel_launch(void* const* d_in, const int* in_sizes, int n_in,
                              void* d_out, int out_size, void* d_ws, size_t ws_size,
                              hipStream_t stream) {
    const float* yi = (const float*)d_in[0];
    const float* yt = (const float*)d_in[1];

    char* ws = (char*)d_ws;
    const size_t szYF = (size_t)NPTS * DIM * sizeof(float);     // 4 MB
    const size_t szYH = (size_t)NPTS * DIM * sizeof(_Float16);  // 2 MB
    const size_t szSQ = (size_t)NPTS * sizeof(float);
    const size_t szTI = (size_t)NPTS * KNN * sizeof(int);

    float*    yinF   = (float*)(ws);
    float*    yitF   = (float*)(ws + szYF);
    _Float16* yinH   = (_Float16*)(ws + 2 * szYF);
    float*    sqArr  = (float*)(ws + 2 * szYF + szYH);
    int*      topIdx = (int*)(ws + 2 * szYF + szYH + szSQ);
    float*    partial= (float*)(ws + 2 * szYF + szYH + szSQ + szTI);

    normalize_kernel<<<NPTS / 8, 256, 0, stream>>>(yi, yt, yinF, yitF, yinH, sqArr);
    topk_kernel<<<NPTS / 16, 512, 0, stream>>>(yinH, sqArr, topIdx);
    loss_kernel<<<NPTS / 8, 256, 0, stream>>>(yinF, yitF, topIdx, partial);
    finalize_kernel<<<1, 512, 0, stream>>>(partial, (float*)d_out);
}
